// AttentionDecoder_12773232738566
// MI455X (gfx1250) — matrix operand order repaired and checked by reading, not compiled, not run on hardware
//
#include <hip/hip_runtime.h>

// ---------- types (trivial ext vectors so they can live in unions) ----------
typedef __attribute__((ext_vector_type(16))) __bf16        v16bf;
typedef __attribute__((ext_vector_type(8)))  float         v8f;
typedef __attribute__((ext_vector_type(4)))  unsigned int  u32x4;
typedef int v4i __attribute__((vector_size(16)));
typedef __attribute__((address_space(1))) v4i as1_v4i;     // global
typedef __attribute__((address_space(3))) v4i as3_v4i;     // LDS

union Frag16 { v16bf v; u32x4 q[2]; };
union Half8  { u32x4 q; unsigned short h[8]; };

__device__ __forceinline__ unsigned short f2bf(float f) {
    unsigned int u = __float_as_uint(f);
    unsigned int r = (u + 0x7fffu + ((u >> 16) & 1u)) >> 16;   // RNE
    return (unsigned short)r;
}

// ---- CDNA5 async global->LDS copy (ASYNCcnt) with synchronous fallback ----
__device__ __forceinline__ void async_ld16(const unsigned short* g, unsigned short* l) {
#if __has_builtin(__builtin_amdgcn_global_load_async_to_lds_b128)
    __builtin_amdgcn_global_load_async_to_lds_b128(
        (as1_v4i*)g, (as3_v4i*)l, 0, 0);
#else
    *(u32x4*)l = *(const u32x4*)g;
#endif
}
__device__ __forceinline__ void wait_async_0() {
#if __has_builtin(__builtin_amdgcn_s_wait_asynccnt)
    __builtin_amdgcn_s_wait_asynccnt(0);
#else
    asm volatile("s_wait_asynccnt 0" ::: "memory");
#endif
}
__device__ __forceinline__ void wait_async_4() {
#if __has_builtin(__builtin_amdgcn_s_wait_asynccnt)
    __builtin_amdgcn_s_wait_asynccnt(4);
#else
    asm volatile("s_wait_asynccnt 4" ::: "memory");
#endif
}

#define GF_RELU   1
#define GF_RESID  2
#define GF_OUTBF  4

// ---------------------------------------------------------------------------
// GEMM: C[M,N] = act(A[M,K](bf16) x Wt[N,K](bf16, pre-transposed) + bias)
//       (+ fp32 residual). Block tile 128x128, 8 waves as 4(M)x2(N),
//       wave tile 32x64 (2x4 WMMA), K-step 32, double-buffered async staging.
// ---------------------------------------------------------------------------
__global__ __launch_bounds__(256) void k_gemm(
    const unsigned short* __restrict__ A, const unsigned short* __restrict__ Wt,
    const float* __restrict__ bias, const float* __restrict__ resid,
    unsigned short* __restrict__ outb, float* __restrict__ outf,
    int M, int N, int K, int flags)
{
    __shared__ __attribute__((aligned(16))) unsigned short As[2][128 * 40];
    __shared__ __attribute__((aligned(16))) unsigned short Bs[2][128 * 40];

    const int tid  = threadIdx.x;
    const int lane = tid & 31;
    const int wave = tid >> 5;
    const int wm   = wave >> 1;          // 0..3  (M direction)
    const int wn   = wave & 1;           // 0..1  (N direction)
    const int bm   = blockIdx.y * 128;
    const int bn   = blockIdx.x * 128;
    const int l16  = lane & 15;
    const int hl   = (lane < 16) ? 0 : 1;

    // staging assignment: each thread copies one 16-half chunk of A and of B
    const int srow = tid >> 1;           // 0..127
    const int sc16 = (tid & 1) * 16;     // 0 or 16

    v8f acc[2][4];
#pragma unroll
    for (int i = 0; i < 2; i++)
#pragma unroll
        for (int j = 0; j < 4; j++)
#pragma unroll
            for (int e = 0; e < 8; e++) acc[i][j][e] = 0.0f;

    auto issue = [&](int buf, int k0) {
        const unsigned short* ga = A  + (size_t)(bm + srow) * K + k0 + sc16;
        async_ld16(ga,     &As[buf][srow * 40 + sc16]);
        async_ld16(ga + 8, &As[buf][srow * 40 + sc16 + 8]);
        const unsigned short* gb = Wt + (size_t)(bn + srow) * K + k0 + sc16;
        async_ld16(gb,     &Bs[buf][srow * 40 + sc16]);
        async_ld16(gb + 8, &Bs[buf][srow * 40 + sc16 + 8]);
    };

    issue(0, 0);
    int cur = 0;
    for (int k0 = 0; k0 < K; k0 += 32) {
        const bool more = (k0 + 32 < K);
        if (more) { issue(cur ^ 1, k0 + 32); wait_async_4(); }
        else      { wait_async_0(); }
        __syncthreads();

        Frag16 af[2], bf[4];
#pragma unroll
        for (int mi = 0; mi < 2; mi++) {
            int r  = wm * 32 + mi * 16 + l16;
            int ko = hl * 8;                              // A frag: K{0..7|8..15}, +16
            af[mi].q[0] = *(const u32x4*)(&As[cur][r * 40 + ko]);
            af[mi].q[1] = *(const u32x4*)(&As[cur][r * 40 + ko + 16]);
        }
#pragma unroll
        for (int ni = 0; ni < 4; ni++) {
            int n  = wn * 64 + ni * 16 + l16;
            int ko = hl * 8;                              // B frag: K{0..7|8..15}, +16
            bf[ni].q[0] = *(const u32x4*)(&Bs[cur][n * 40 + ko]);
            bf[ni].q[1] = *(const u32x4*)(&Bs[cur][n * 40 + ko + 16]);
        }
#pragma unroll
        for (int mi = 0; mi < 2; mi++)
#pragma unroll
            for (int ni = 0; ni < 4; ni++)
                acc[mi][ni] = __builtin_amdgcn_wmma_f32_16x16x32_bf16(
                    false, af[mi].v, false, bf[ni].v, (short)0, acc[mi][ni], false, false);
        __syncthreads();
        cur ^= 1;
    }

    // ---- fused epilogue ----
#pragma unroll
    for (int mi = 0; mi < 2; mi++) {
#pragma unroll
        for (int ni = 0; ni < 4; ni++) {
            int col  = bn + wn * 64 + ni * 16 + l16;
            float bv = bias ? bias[col] : 0.0f;
#pragma unroll
            for (int r = 0; r < 8; r++) {
                int row = bm + wm * 32 + mi * 16 + r + hl * 8;
                float v = acc[mi][ni][r] + bv;
                if (flags & GF_RELU) v = v > 0.0f ? v : 0.0f;
                size_t idx = (size_t)row * N + col;
                if (flags & GF_RESID)      outf[idx] = resid[idx] + v;
                else if (flags & GF_OUTBF) outb[idx] = f2bf(v);
                else                       outf[idx] = v;
            }
        }
    }
}

// ---------------------------------------------------------------------------
// Flash attention: one (b, h, 64-query block) per 128-thread block (4 waves),
// each wave owns 16 query rows. Key blocks of 32, causal.
// ---------------------------------------------------------------------------
__global__ __launch_bounds__(128) void k_attn(
    const unsigned short* __restrict__ qkv, unsigned short* __restrict__ aout,
    int B, int S, int D, int H)
{
    __shared__ __attribute__((aligned(16))) unsigned short Ks[32 * 72];       // [key][hd]
    __shared__ __attribute__((aligned(16))) unsigned short Vt[64 * 40];       // [hd][key]
    __shared__ __attribute__((aligned(16))) unsigned short Ps[4 * 16 * 40];   // per-wave P

    const int tid  = threadIdx.x;
    const int lane = tid & 31;
    const int wave = tid >> 5;
    const int l16  = lane & 15;
    const int hl   = (lane < 16) ? 0 : 1;

    const int qblk = blockIdx.x;
    const int bh   = blockIdx.y;
    const int b    = bh / H;
    const int h    = bh % H;
    const size_t rstr = (size_t)3 * D;
    const int qrow0 = qblk * 64 + wave * 16;

    // Q fragments (hd = 64 -> two K-steps of 32) stay in registers
    Frag16 qa[2];
    {
        const unsigned short* p = qkv + ((size_t)(b * S + qrow0 + l16)) * rstr + h * 64;
#pragma unroll
        for (int s = 0; s < 2; s++) {
            int ko = s * 32 + hl * 8;
            qa[s].q[0] = *(const u32x4*)(p + ko);
            qa[s].q[1] = *(const u32x4*)(p + ko + 16);
        }
    }

    float mrow[8], srow[8];
    v8f o[4];
#pragma unroll
    for (int r = 0; r < 8; r++) { mrow[r] = -1e30f; srow[r] = 0.0f; }
#pragma unroll
    for (int n = 0; n < 4; n++)
#pragma unroll
        for (int r = 0; r < 8; r++) o[n][r] = 0.0f;

    const int nkb = (qblk * 64 + 64) / 32;
    for (int kb = 0; kb < nkb; kb++) {
        // ---- stage K via async copies (natural layout), V via transpose ----
#pragma unroll
        for (int i = 0; i < 2; i++) {
            int c   = tid * 2 + i;       // 0..255
            int key = c >> 3;            // 0..31
            int c8  = (c & 7) * 8;       // 0..56
            const unsigned short* kp =
                qkv + ((size_t)(b * S + kb * 32 + key)) * rstr + D + h * 64 + c8;
            async_ld16(kp, Ks + key * 72 + c8);
            Half8 hv; hv.q = *(const u32x4*)(kp + D);     // V = K + D
#pragma unroll
            for (int e = 0; e < 8; e++) Vt[(c8 + e) * 40 + key] = hv.h[e];
        }
        wait_async_0();
        __syncthreads();

        // ---- scores = Q K^T (two 16-key tiles, hd contracted in 2 steps) ----
        v8f sc[2];
#pragma unroll
        for (int kt = 0; kt < 2; kt++)
#pragma unroll
            for (int e = 0; e < 8; e++) sc[kt][e] = 0.0f;
#pragma unroll
        for (int s = 0; s < 2; s++) {
#pragma unroll
            for (int kt = 0; kt < 2; kt++) {
                Frag16 kf;
                const unsigned short* kp = Ks + (kt * 16 + l16) * 72 + s * 32 + hl * 8;
                kf.q[0] = *(const u32x4*)kp;
                kf.q[1] = *(const u32x4*)(kp + 16);
                sc[kt] = __builtin_amdgcn_wmma_f32_16x16x32_bf16(
                    false, qa[s].v, false, kf.v, (short)0, sc[kt], false, false);
            }
        }

        // ---- causal mask + online softmax (row = r + hl*8, col = l16) ----
        float pr[2][8];
#pragma unroll
        for (int r = 0; r < 8; r++) {
            int q = qrow0 + r + hl * 8;
            float rowmax = -1e30f;
#pragma unroll
            for (int kt = 0; kt < 2; kt++) {
                int kcol = kb * 32 + kt * 16 + l16;
                float f = sc[kt][r] * 0.125f;             // 1/sqrt(64)
                if (kcol > q) f = -10000.0f;
                pr[kt][r] = f;
                rowmax = fmaxf(rowmax, f);
            }
#pragma unroll
            for (int m = 8; m >= 1; m >>= 1)
                rowmax = fmaxf(rowmax, __shfl_xor(rowmax, m, 32));
            float newm  = fmaxf(mrow[r], rowmax);
            float alpha = __expf(mrow[r] - newm);
            float rsum  = 0.0f;
#pragma unroll
            for (int kt = 0; kt < 2; kt++) {
                float p = __expf(pr[kt][r] - newm);
                pr[kt][r] = p;
                rsum += p;
            }
#pragma unroll
            for (int m = 8; m >= 1; m >>= 1) rsum += __shfl_xor(rsum, m, 32);
            srow[r] = srow[r] * alpha + rsum;
            mrow[r] = newm;
#pragma unroll
            for (int n = 0; n < 4; n++) o[n][r] = o[n][r] * alpha;
        }

        // ---- transpose P through per-wave LDS patch ----
        unsigned short* Pw = Ps + wave * 16 * 40;
#pragma unroll
        for (int kt = 0; kt < 2; kt++)
#pragma unroll
            for (int r = 0; r < 8; r++)
                Pw[(r + hl * 8) * 40 + kt * 16 + l16] = f2bf(pr[kt][r]);
        asm volatile("s_wait_dscnt 0" ::: "memory");

        Frag16 pa;
        {
            const unsigned short* pp = Pw + l16 * 40 + hl * 8;
            pa.q[0] = *(const u32x4*)pp;
            pa.q[1] = *(const u32x4*)(pp + 16);
        }
        // ---- O += P x V ----
#pragma unroll
        for (int n = 0; n < 4; n++) {
            Frag16 vf;
            const unsigned short* vp = Vt + (n * 16 + l16) * 40 + hl * 8;
            vf.q[0] = *(const u32x4*)vp;
            vf.q[1] = *(const u32x4*)(vp + 16);
            o[n] = __builtin_amdgcn_wmma_f32_16x16x32_bf16(
                false, pa.v, false, vf.v, (short)0, o[n], false, false);
        }
        __syncthreads();
    }

    // ---- write merged-head output, bf16 ----
#pragma unroll
    for (int n = 0; n < 4; n++) {
        int col = h * 64 + n * 16 + l16;
#pragma unroll
        for (int r = 0; r < 8; r++) {
            int q = qrow0 + r + hl * 8;
            aout[((size_t)(b * S + q)) * D + col] = f2bf(o[n][r] / srow[r]);
        }
    }
}

// ---------------------------------------------------------------------------
// LayerNorm: one row per 256-thread block, fp32 in -> bf16 out
// ---------------------------------------------------------------------------
__global__ __launch_bounds__(256) void k_ln(
    const float* __restrict__ x, const float* __restrict__ g,
    const float* __restrict__ bta, unsigned short* __restrict__ out, int D)
{
    __shared__ float red[16];
    const int row = blockIdx.x;
    const int tid = threadIdx.x;
    const float* xr = x + (size_t)row * D;

    float s = 0.0f, ss = 0.0f;
    for (int i = tid; i < D; i += 256) { float v = xr[i]; s += v; ss += v * v; }
#pragma unroll
    for (int m = 16; m >= 1; m >>= 1) { s += __shfl_xor(s, m, 32); ss += __shfl_xor(ss, m, 32); }
    if ((tid & 31) == 0) { red[tid >> 5] = s; red[8 + (tid >> 5)] = ss; }
    __syncthreads();
    float ts = 0.0f, tss = 0.0f;
#pragma unroll
    for (int w = 0; w < 8; w++) { ts += red[w]; tss += red[8 + w]; }
    float mean = ts / (float)D;
    float var  = tss / (float)D - mean * mean;
    float rstd = rsqrtf(var + 1e-5f);
    for (int i = tid; i < D; i += 256)
        out[(size_t)row * D + i] = f2bf((xr[i] - mean) * rstd * g[i] + bta[i]);
}

// ---------------------------------------------------------------------------
// Tiled transpose + fp32->bf16 convert: W[K][N] fp32  ->  Wt[N][K] bf16
// ---------------------------------------------------------------------------
__global__ __launch_bounds__(256) void k_cvt_t(const float* __restrict__ W,
                                               unsigned short* __restrict__ Wt,
                                               int K, int N)
{
    __shared__ unsigned short T[32][33];
    const int bk = blockIdx.y * 32;
    const int bn = blockIdx.x * 32;
    const int tid = threadIdx.x;
    {
        int kr = tid >> 3;             // 0..31 (k row)
        int n4 = (tid & 7) * 4;        // 0..28
        const float* src = W + (size_t)(bk + kr) * N + bn + n4;
#pragma unroll
        for (int e = 0; e < 4; e++) T[n4 + e][kr] = f2bf(src[e]);
    }
    __syncthreads();
    {
        int nr = tid >> 3;             // 0..31 (n row)
        int k4 = (tid & 7) * 4;        // 0..28
        unsigned short* dst = Wt + (size_t)(bn + nr) * K + bk + k4;
#pragma unroll
        for (int e = 0; e < 4; e++) dst[e] = T[nr][k4 + e];
    }
}

// ---------------------------------------------------------------------------
__global__ __launch_bounds__(256) void k_copyf(const float* __restrict__ in,
                                               float* __restrict__ out, long n)
{
    long i = (long)blockIdx.x * 256 + threadIdx.x;
    long st = (long)gridDim.x * 256;
    for (; i < n; i += st) out[i] = in[i];
}

// ---------------------------------------------------------------------------
extern "C" void kernel_launch(void* const* d_in, const int* in_sizes, int n_in,
                              void* d_out, int out_size, void* d_ws, size_t ws_size,
                              hipStream_t stream)
{
    (void)in_sizes; (void)n_in; (void)out_size; (void)ws_size;
    const int B = 4, S = 1024, D = 1024, L = 4, H = 16;
    const int BS = B * S;

    const float* x     = (const float*)d_in[0];
    const float* ln1g  = (const float*)d_in[1];
    const float* ln1b  = (const float*)d_in[2];
    const float* wattn = (const float*)d_in[3];
    const float* battn = (const float*)d_in[4];
    const float* wproj = (const float*)d_in[5];
    const float* bproj = (const float*)d_in[6];
    const float* ln2g  = (const float*)d_in[7];
    const float* ln2b  = (const float*)d_in[8];
    const float* wfc   = (const float*)d_in[9];
    const float* bfc   = (const float*)d_in[10];
    const float* wfcp  = (const float*)d_in[11];
    const float* bfcp  = (const float*)d_in[12];
    const float* lnfg  = (const float*)d_in[13];
    const float* lnfb  = (const float*)d_in[14];
    const float* wf    = (const float*)d_in[15];
    const float* bfin  = (const float*)d_in[16];

    char* ws = (char*)d_ws;
    size_t off = 0;
    auto alloc = [&](size_t bytes) -> char* {
        char* p = ws + off;
        off += (bytes + 255) & ~(size_t)255;
        return p;
    };
    float*          xbuf  = (float*)alloc((size_t)BS * D * 4);
    unsigned short* hbuf  = (unsigned short*)alloc((size_t)BS * D * 2);
    unsigned short* qkvb  = (unsigned short*)alloc((size_t)BS * 3 * D * 2);
    unsigned short* abuf  = (unsigned short*)alloc((size_t)BS * D * 2);
    unsigned short* mbuf  = (unsigned short*)alloc((size_t)BS * 4 * D * 2);
    unsigned short* wattnb = (unsigned short*)alloc((size_t)L * D * 3 * D * 2);  // [3D][D] per layer
    unsigned short* wprojb = (unsigned short*)alloc((size_t)L * D * D * 2);      // [D][D]
    unsigned short* wfcb   = (unsigned short*)alloc((size_t)L * D * 4 * D * 2);  // [4D][D]
    unsigned short* wfcpb  = (unsigned short*)alloc((size_t)L * 4 * D * D * 2);  // [D][4D]
    unsigned short* wfb    = (unsigned short*)alloc((size_t)D * D * 2);          // [D][D]

    // weights: fp32 -> bf16 + transpose to [N][K] (once per launch; deterministic)
    for (int i = 0; i < L; i++) {
        k_cvt_t<<<dim3(3 * D / 32, D / 32), 256, 0, stream>>>(
            wattn + (size_t)i * D * 3 * D, wattnb + (size_t)i * D * 3 * D, D, 3 * D);
        k_cvt_t<<<dim3(D / 32, D / 32), 256, 0, stream>>>(
            wproj + (size_t)i * D * D, wprojb + (size_t)i * D * D, D, D);
        k_cvt_t<<<dim3(4 * D / 32, D / 32), 256, 0, stream>>>(
            wfc + (size_t)i * D * 4 * D, wfcb + (size_t)i * D * 4 * D, D, 4 * D);
        k_cvt_t<<<dim3(D / 32, 4 * D / 32), 256, 0, stream>>>(
            wfcp + (size_t)i * 4 * D * D, wfcpb + (size_t)i * 4 * D * D, 4 * D, D);
    }
    k_cvt_t<<<dim3(D / 32, D / 32), 256, 0, stream>>>(wf, wfb, D, D);
    k_copyf<<<2048, 256, 0, stream>>>(x, xbuf, (long)BS * D);

    for (int i = 0; i < L; i++) {
        k_ln<<<BS, 256, 0, stream>>>(xbuf, ln1g + (size_t)i * D, ln1b + (size_t)i * D, hbuf, D);
        k_gemm<<<dim3(3 * D / 128, BS / 128), 256, 0, stream>>>(
            hbuf, wattnb + (size_t)i * D * 3 * D, battn + (size_t)i * 3 * D,
            nullptr, qkvb, nullptr, BS, 3 * D, D, GF_OUTBF);
        k_attn<<<dim3(S / 64, B * H), 128, 0, stream>>>(qkvb, abuf, B, S, D, H);
        k_gemm<<<dim3(D / 128, BS / 128), 256, 0, stream>>>(
            abuf, wprojb + (size_t)i * D * D, bproj + (size_t)i * D,
            xbuf, nullptr, xbuf, BS, D, D, GF_RESID);
        k_ln<<<BS, 256, 0, stream>>>(xbuf, ln2g + (size_t)i * D, ln2b + (size_t)i * D, hbuf, D);
        k_gemm<<<dim3(4 * D / 128, BS / 128), 256, 0, stream>>>(
            hbuf, wfcb + (size_t)i * D * 4 * D, bfc + (size_t)i * 4 * D,
            nullptr, mbuf, nullptr, BS, 4 * D, D, GF_OUTBF | GF_RELU);
        k_gemm<<<dim3(D / 128, BS / 128), 256, 0, stream>>>(
            mbuf, wfcpb + (size_t)i * 4 * D * D, bfcp + (size_t)i * D,
            xbuf, nullptr, xbuf, BS, D, 4 * D, GF_RESID);
    }
    k_ln<<<BS, 256, 0, stream>>>(xbuf, lnfg, lnfb, hbuf, D);
    k_gemm<<<dim3(D / 128, BS / 128), 256, 0, stream>>>(
        hbuf, wfb, bfin, nullptr, nullptr, (float*)d_out, BS, D, D, 0);
}
